// EfficientVQBlock_11407433138592
// MI455X (gfx1250) — matrix operand order repaired and checked by reading, not compiled, not run on hardware
//
#include <hip/hip_runtime.h>
#include <hip/hip_bf16.h>

// ---------------------------------------------------------------------------
// EfficientVQBlock for MI455X (gfx1250, wave32, WMMA bf16 16x16x32)
//   - weights pre-converted to bf16 in global (read once per tile, half traffic)
//   - A tiles staged with global_load_async_to_lds_b128 (ASYNCcnt path)
//   - B tiles software-pipelined through VGPRs with packed f32->bf16 cvt
//   - fragment-contiguous LDS layouts -> ds_load_b128 fragment loads,
//     conflict-free banking
//   - WMMA-based VQ code search with in-register argmin reduction
// ---------------------------------------------------------------------------

typedef __attribute__((ext_vector_type(16))) __bf16 v16bf;
typedef __attribute__((ext_vector_type(8)))  __bf16 v8bf;
typedef __attribute__((ext_vector_type(4)))  __bf16 v4bf;
typedef __attribute__((ext_vector_type(8)))  float  v8f;
typedef __attribute__((ext_vector_type(4)))  int    v4i;

#define BN_EPS 1e-5f

#define GLOBAL_AS __attribute__((address_space(1)))
#define LDS_AS    __attribute__((address_space(3)))

#if defined(__has_builtin)
#if __has_builtin(__builtin_amdgcn_global_load_async_to_lds_b128)
#define HAVE_ASYNC_LDS 1
#endif
#endif
#ifndef HAVE_ASYNC_LDS
#define HAVE_ASYNC_LDS 0
#endif

__device__ __forceinline__ __bf16 f2bf(float x) { return (__bf16)x; }
__device__ __forceinline__ float hswish_f(float x) {
  return x * fminf(fmaxf(x + 3.0f, 0.0f), 6.0f) * (1.0f / 6.0f);
}

__device__ __forceinline__ void wait_async0() {
#if HAVE_ASYNC_LDS
#if __has_builtin(__builtin_amdgcn_s_wait_asynccnt)
  __builtin_amdgcn_s_wait_asynccnt(0);
#else
  asm volatile("s_wait_asynccnt 0x0" ::: "memory");
#endif
#endif
}

// ---------------------------------------------------------------------------
// Tiled WMMA GEMM:  C[M,N] = A[M,K] * B[K,N]
//   A: bf16 row-major (pre-converted weights)   B,C: f32 row-major
// grid = (N/64, M/64, batch); block = 128 threads (4 waves, 2x2 wave tiles).
// EPI: 0 = none, 1 = bias + hswish, 2 = batchnorm, 3 = batchnorm + residual
//
// LDS layouts (bf16, row stride 40 elements = 80B, 16B aligned rows):
//   Abuf[r][0..31]  : K-groups of 8 in natural order (contiguous k) -> a lane's 16-element
//                     WMMA A fragment is contiguous at r*40 + hi*16.
//   Bbuf[col][0..31]: K-major (transposed)   -> B fragment contiguous at
//                     col*40 + hi*16.
// ---------------------------------------------------------------------------
template <int EPI>
__global__ __launch_bounds__(128)
void gemm_wmma(const __bf16* __restrict__ A, const float* __restrict__ Bmat,
               float* __restrict__ C, int M, int N, int K,
               long strideB, long strideC, long strideR,
               const float* __restrict__ bias,
               const float* __restrict__ bn_g, const float* __restrict__ bn_b,
               const float* __restrict__ bn_m, const float* __restrict__ bn_v,
               const float* __restrict__ R) {
  __shared__ __align__(16) __bf16 Abuf[2][64 * 40];
  __shared__ __align__(16) __bf16 Bbuf[2][64 * 40];

  const int tid = threadIdx.x;
  const int m0 = blockIdx.y * 64;
  const int n0 = blockIdx.x * 64;
  const float* Bp = Bmat + (long)blockIdx.z * strideB;
  float*       Cp = C    + (long)blockIdx.z * strideC;
  const float* Rp = (EPI == 3) ? (R + (long)blockIdx.z * strideR) : nullptr;

  const int wv = tid >> 5;
  const int lane = tid & 31;
  const int l  = lane & 15;
  const int hi = lane >> 4;
  const int rowbase = (wv >> 1) * 32;
  const int colbase = (wv & 1) * 32;

  v8f acc[2][2] = {};
  float breg[4][4];

  const int nch = K >> 5;

  // ---- stage A chunk: 64 rows x 4 16B-groups, stored in natural k order.
  //      async path: DMA straight into LDS (no VGPR round trip, ASYNCcnt).
  auto stage_A = [&](int buf, int kc) {
#pragma unroll
    for (int it = 0; it < 2; ++it) {
      int q = tid + it * 128;                 // 256 chunks of 16B
      int r = q >> 2, g = q & 3;
      int ng = g;                             // 0,1,2,3 -> 0,1,2,3 (contiguous k)
      const __bf16* gsrc = &A[(long)(m0 + r) * K + kc + g * 8];
      __bf16* ldst = &Abuf[buf][r * 40 + ng * 8];
#if HAVE_ASYNC_LDS
      __builtin_amdgcn_global_load_async_to_lds_b128(
          (GLOBAL_AS v4i*)gsrc, (LDS_AS v4i*)ldst, 0, 0);
#else
      *(v8bf*)ldst = *(const v8bf*)gsrc;
#endif
    }
  };

  // ---- B: issue global f32 loads into registers (software pipeline) ----
  auto load_B = [&](int kc) {
#pragma unroll
    for (int it = 0; it < 4; ++it) {
      int q = tid + it * 128;                 // 64 cols x 8 k-quads
      int col = q & 63, kb = (q >> 6) * 4;
#pragma unroll
      for (int j = 0; j < 4; ++j)
        breg[it][j] = Bp[(long)(kc + kb + j) * N + n0 + col];
    }
  };
  // ---- B: convert + store to LDS buffer ----
  auto store_B = [&](int buf) {
#pragma unroll
    for (int it = 0; it < 4; ++it) {
      int q = tid + it * 128;
      int col = q & 63, kb = (q >> 6) * 4;
      v4bf v;
#pragma unroll
      for (int j = 0; j < 4; ++j) v[j] = f2bf(breg[it][j]);
      *(v4bf*)&Bbuf[buf][col * 40 + kb] = v;
    }
  };

  stage_A(0, 0);
  load_B(0);
  store_B(0);
  wait_async0();
  __syncthreads();

  for (int ch = 0; ch < nch; ++ch) {
    const int cur = ch & 1;
    const bool hasNext = (ch + 1) < nch;
    if (hasNext) {
      stage_A(cur ^ 1, (ch + 1) << 5);       // async DMA into other buffer
      load_B((ch + 1) << 5);                 // issue global loads early
    }
    if (ch + 2 < nch) {                      // warm L2 two chunks ahead
      __builtin_prefetch(&A[(long)(m0 + (tid >> 1)) * K + ((ch + 2) << 5)], 0, 0);
      __builtin_prefetch(&Bp[(long)(((ch + 2) << 5) + (tid >> 6)) * N + n0 + (tid & 63)], 0, 0);
    }

    // fragment loads: two ds_load_b128 per fragment, conflict-free banking
    v16bf af[2], bfr[2];
#pragma unroll
    for (int i = 0; i < 2; ++i) {
      const __bf16* ap = &Abuf[cur][(rowbase + i * 16 + l) * 40 + hi * 16];
      v8bf a0 = *(const v8bf*)ap;
      v8bf a1 = *(const v8bf*)(ap + 8);
      af[i] = __builtin_shufflevector(a0, a1, 0, 1, 2, 3, 4, 5, 6, 7,
                                      8, 9, 10, 11, 12, 13, 14, 15);
    }
#pragma unroll
    for (int j = 0; j < 2; ++j) {
      const __bf16* bp = &Bbuf[cur][(colbase + j * 16 + l) * 40 + hi * 16];
      v8bf b0 = *(const v8bf*)bp;
      v8bf b1 = *(const v8bf*)(bp + 8);
      bfr[j] = __builtin_shufflevector(b0, b1, 0, 1, 2, 3, 4, 5, 6, 7,
                                       8, 9, 10, 11, 12, 13, 14, 15);
    }

#pragma unroll
    for (int i = 0; i < 2; ++i)
#pragma unroll
      for (int j = 0; j < 2; ++j)
        acc[i][j] = __builtin_amdgcn_wmma_f32_16x16x32_bf16(
            false, af[i], false, bfr[j], (short)0, acc[i][j], false, false);

    if (hasNext) store_B(cur ^ 1);
    wait_async0();                           // next-buffer DMA landed
    __syncthreads();
  }

  // Epilogue. C layout: VGPR r of lane L -> (row r + 8*(L>>4), col L&15).
#pragma unroll
  for (int i = 0; i < 2; ++i) {
#pragma unroll
    for (int j = 0; j < 2; ++j) {
#pragma unroll
      for (int r = 0; r < 8; ++r) {
        int row = m0 + rowbase + i * 16 + r + hi * 8;
        int col = n0 + colbase + j * 16 + l;
        float v = acc[i][j][r];
        if (EPI == 1) {
          v = hswish_f(v + bias[row]);
        } else if (EPI >= 2) {
          float s = bn_g[row] * rsqrtf(bn_v[row] + BN_EPS);
          v = (v - bn_m[row]) * s + bn_b[row];
          if (EPI == 3) v += Rp[(long)row * N + col];
        }
        Cp[(long)row * N + col] = v;
      }
    }
  }
}

// ---------------------------------------------------------------------------
// VQ code search: scores = z . c_k via one bf16 WMMA (K = D = 32), then
// argmin_k (||c_k||^2/2 - z.c_k) with in-register cross-lane reduction.
// grid = (N/64, B*m); block = 128 (each wave owns 16 spatial rows).
// ---------------------------------------------------------------------------
__global__ __launch_bounds__(128)
void vq_codes_kernel(const float* __restrict__ ms, const float* __restrict__ cb,
                     const float* __restrict__ cn2, int* __restrict__ idx,
                     int part) {
  const int tid = threadIdx.x;
  const int wv = tid >> 5;
  const int lane = tid & 31;
  const int l  = lane & 15;
  const int hi = lane >> 4;
  const int nbase = blockIdx.x * 64 + wv * 16;
  const int b  = blockIdx.y >> 4;
  const int mi = blockIdx.y & 15;
  const int chbase = mi * 96 + part * 32;

  const float* zp = ms + ((long)b * 1536 + chbase) * 1024 + nbase + l;
  v16bf a;
#pragma unroll
  for (int e = 0; e < 16; ++e) {
    int kk = e + (hi << 4);
    a[e] = f2bf(zp[(long)kk * 1024]);
  }

  float best[8];
  int   bidx[8];
#pragma unroll
  for (int r = 0; r < 8; ++r) { best[r] = 3.0e38f; bidx[r] = 0; }

  for (int ct = 0; ct < 16; ++ct) {          // 16 tiles of 16 codes = 256
    v16bf bfr;
    const float* cp = cb + (ct * 16 + l) * 32 + hi * 16;
#pragma unroll
    for (int e = 0; e < 16; ++e) bfr[e] = f2bf(cp[e]);
    v8f s = {};
    s = __builtin_amdgcn_wmma_f32_16x16x32_bf16(false, a, false, bfr, (short)0,
                                                s, false, false);
    int code = ct * 16 + l;
    float c2h = 0.5f * cn2[code];
#pragma unroll
    for (int r = 0; r < 8; ++r) {
      float key = c2h - s[r];
      if (key < best[r] || (key == best[r] && code < bidx[r])) {
        best[r] = key; bidx[r] = code;
      }
    }
  }
#pragma unroll
  for (int r = 0; r < 8; ++r) {
#pragma unroll
    for (int off = 1; off < 16; off <<= 1) {
      float ov = __shfl_xor(best[r], off, 32);
      int   oi = __shfl_xor(bidx[r], off, 32);
      if (ov < best[r] || (ov == best[r] && oi < bidx[r])) {
        best[r] = ov; bidx[r] = oi;
      }
    }
    if (l == 0)
      idx[((long)b * 16 + mi) * 1024 + nbase + r + 8 * hi] = bidx[r];
  }
}

// ---------------------------------------------------------------------------
// Small helper kernels (bandwidth-bound; plain VALU)
// ---------------------------------------------------------------------------
__global__ void f32_to_bf16_kernel(const float* __restrict__ in,
                                   __bf16* __restrict__ out, long n4) {
  long t = (long)blockIdx.x * blockDim.x + threadIdx.x;
  if (t >= n4) return;
  float4 f = *(const float4*)&in[t * 4];
  v4bf v;
  v[0] = f2bf(f.x); v[1] = f2bf(f.y); v[2] = f2bf(f.z); v[3] = f2bf(f.w);
  *(v4bf*)&out[t * 4] = v;
}

__global__ void cn2_kernel(const float* __restrict__ cb, float* __restrict__ cn2) {
  int k = threadIdx.x;  // 256 threads
  float s = 0.f;
#pragma unroll
  for (int d = 0; d < 32; ++d) { float t = cb[k * 32 + d]; s += t * t; }
  cn2[k] = s;
}

__global__ void dw5_kernel(const float* __restrict__ ms,
                           const float* __restrict__ w,
                           float* __restrict__ out) {
  long t = (long)blockIdx.x * blockDim.x + threadIdx.x;
  if (t >= (long)16 * 768 * 1024) return;
  int n = (int)(t & 1023);
  int c = (int)((t >> 10) % 768);
  int b = (int)(t / ((long)768 * 1024));
  int h = n >> 5, x = n & 31;
  const float* ip = ms + ((long)b * 1536 + c) * 1024;
  const float* wp = w + c * 25;
  float s = 0.f;
#pragma unroll
  for (int kh = 0; kh < 5; ++kh) {
    int ih = h + kh - 2;
    if (ih < 0 || ih >= 32) continue;
#pragma unroll
    for (int kw = 0; kw < 5; ++kw) {
      int iw = x + kw - 2;
      if (iw < 0 || iw >= 32) continue;
      s += ip[ih * 32 + iw] * wp[kh * 5 + kw];
    }
  }
  out[t] = s;
}

__global__ void grouped_pw_kernel(const float* __restrict__ in,
                                  const float* __restrict__ w,
                                  float* __restrict__ ms) {
  long t = (long)blockIdx.x * blockDim.x + threadIdx.x;
  if (t >= (long)16 * 768 * 1024) return;
  int n = (int)(t & 1023);
  int o = (int)((t >> 10) % 768);
  int b = (int)(t / ((long)768 * 1024));
  const float* ip = in + ((long)b * 768 + ((o >> 5) << 5)) * 1024 + n;
  const float* wp = w + o * 32;
  float s = 0.f;
#pragma unroll
  for (int ci = 0; ci < 32; ++ci) s += ip[(long)ci * 1024] * wp[ci];
  ms[((long)b * 1536 + 768 + o) * 1024 + n] = s;
}

__global__ void vq_scatter_kernel(const float* __restrict__ ms,
                                  const int* __restrict__ idxk,
                                  float* __restrict__ sums,
                                  float* __restrict__ counts) {
  long t = (long)blockIdx.x * blockDim.x + threadIdx.x;
  if (t >= (long)16 * 16 * 1024) return;
  int n  = (int)(t & 1023);
  int mi = (int)((t >> 10) & 15);
  int b  = (int)(t >> 14);
  int kc = idxk[t];
  const float* vp = ms + ((long)b * 1536 + mi * 96 + 64) * 1024 + n;
  float* sp = sums + (((long)b * 16 + mi) * 256 + kc) * 32;
#pragma unroll
  for (int d = 0; d < 32; ++d) atomicAdd(&sp[d], vp[(long)d * 1024]);
  atomicAdd(&counts[((long)b * 16 + mi) * 256 + kc], 1.0f);
}

__global__ void vq_gather_kernel(const int* __restrict__ idxq,
                                 const float* __restrict__ sums,
                                 const float* __restrict__ counts,
                                 float* __restrict__ vqout) {
  long t = (long)blockIdx.x * blockDim.x + threadIdx.x;
  if (t >= (long)16 * 16 * 1024) return;
  int n  = (int)(t & 1023);
  int mi = (int)((t >> 10) & 15);
  int b  = (int)(t >> 14);
  int qc = idxq[t];
  long base = ((long)b * 16 + mi) * 256 + qc;
  float inv = 1.0f / fmaxf(counts[base], 1.0f);
  const float* sp = sums + base * 32;
  float* op = vqout + ((long)b * 512 + mi * 32) * 1024 + n;
#pragma unroll
  for (int d = 0; d < 32; ++d) op[(long)d * 1024] = sp[d] * inv;
}

__global__ void dw3_kernel(const float* __restrict__ in,
                           const float* __restrict__ w,
                           const float* __restrict__ bias,
                           float* __restrict__ out) {
  long t = (long)blockIdx.x * blockDim.x + threadIdx.x;
  if (t >= (long)16 * 1024 * 1024) return;
  int n = (int)(t & 1023);
  int c = (int)((t >> 10) & 1023);
  int b = (int)(t >> 20);
  int h = n >> 5, x = n & 31;
  const float* ip = in + ((long)b * 1024 + c) * 1024;
  const float* wp = w + c * 9;
  float s = bias[c];
#pragma unroll
  for (int kh = 0; kh < 3; ++kh) {
    int ih = h + kh - 1;
    if (ih < 0 || ih >= 32) continue;
#pragma unroll
    for (int kw = 0; kw < 3; ++kw) {
      int iw = x + kw - 1;
      if (iw < 0 || iw >= 32) continue;
      s += ip[ih * 32 + iw] * wp[kh * 3 + kw];
    }
  }
  out[t] = hswish_f(s);
}

// ---------------------------------------------------------------------------
// Launcher
// ---------------------------------------------------------------------------
extern "C" void kernel_launch(void* const* d_in, const int* in_sizes, int n_in,
                              void* d_out, int out_size, void* d_ws,
                              size_t ws_size, hipStream_t stream) {
  const float* x      = (const float*)d_in[0];
  const float* w_qkv  = (const float*)d_in[1];
  const float* w_dw5  = (const float*)d_in[2];
  const float* w_pw_g = (const float*)d_in[3];
  const float* cb     = (const float*)d_in[4];
  const float* w_proj = (const float*)d_in[5];
  const float* bn1_g  = (const float*)d_in[6];
  const float* bn1_b  = (const float*)d_in[7];
  const float* bn1_m  = (const float*)d_in[8];
  const float* bn1_v  = (const float*)d_in[9];
  const float* w_exp  = (const float*)d_in[10];
  const float* b_exp  = (const float*)d_in[11];
  const float* w_dw3  = (const float*)d_in[12];
  const float* b_dw3  = (const float*)d_in[13];
  const float* w_pw   = (const float*)d_in[14];
  const float* bn2_g  = (const float*)d_in[15];
  const float* bn2_b  = (const float*)d_in[16];
  const float* bn2_m  = (const float*)d_in[17];
  const float* bn2_v  = (const float*)d_in[18];

  char* ws = (char*)d_ws;
  size_t off = 0;
  auto alloc = [&](size_t bytes) -> void* {
    off = (off + 255) & ~(size_t)255;
    void* p = ws + off;
    off += bytes;
    return p;
  };

  const long B = 16, N = 1024;
  float* ms    = (float*)alloc((size_t)B * 1536 * N * 4);  // qkv | agg concat
  size_t off_overlay = off;                                // start of VQ slab
  float* agg1  = (float*)alloc((size_t)B * 768 * N * 4);
  int*   idxq  = (int*)  alloc((size_t)B * 16 * N * 4);
  int*   idxk  = (int*)  alloc((size_t)B * 16 * N * 4);
  float* sums  = (float*)alloc((size_t)B * 16 * 256 * 32 * 4);
  float* counts= (float*)alloc((size_t)B * 16 * 256 * 4);
  float* vqout = (float*)alloc((size_t)B * 512 * N * 4);
  float* x1    = (float*)alloc((size_t)B * 256 * N * 4);
  float* cn2   = (float*)alloc(256 * 4);
  __bf16* w_qkv_bf  = (__bf16*)alloc((size_t)768 * 256 * 2);
  __bf16* w_proj_bf = (__bf16*)alloc((size_t)256 * 512 * 2);
  __bf16* w_exp_bf  = (__bf16*)alloc((size_t)1024 * 256 * 2);
  __bf16* w_pw_bf   = (__bf16*)alloc((size_t)256 * 1024 * 2);
  // lifetime-based aliases: ms dead after gather, VQ slab dead after proj
  float* h1 = ms;                                  // [B,1024,N] (67MB <= 100MB)
  float* h2 = (float*)(ws + ((off_overlay + 255) & ~(size_t)255));  // 67MB <= 94MB slab

  // 0) constants: ||c_k||^2 and bf16 weight copies
  cn2_kernel<<<1, 256, 0, stream>>>(cb, cn2);
  f32_to_bf16_kernel<<<(768 * 256 / 4 + 255) / 256, 256, 0, stream>>>(w_qkv, w_qkv_bf, 768L * 256 / 4);
  f32_to_bf16_kernel<<<(256 * 512 / 4 + 255) / 256, 256, 0, stream>>>(w_proj, w_proj_bf, 256L * 512 / 4);
  f32_to_bf16_kernel<<<(1024 * 256 / 4 + 255) / 256, 256, 0, stream>>>(w_exp, w_exp_bf, 1024L * 256 / 4);
  f32_to_bf16_kernel<<<(256 * 1024 / 4 + 255) / 256, 256, 0, stream>>>(w_pw, w_pw_bf, 256L * 1024 / 4);

  // 1) qkv = w_qkv @ x   -> ms[0:768]
  gemm_wmma<0><<<dim3(16, 12, 16), 128, 0, stream>>>(
      w_qkv_bf, x, ms, 768, 1024, 256, 256L * 1024, 1536L * 1024, 0,
      nullptr, nullptr, nullptr, nullptr, nullptr, nullptr);

  // 2) depthwise 5x5 on qkv -> agg1
  {
    long tot = B * 768 * N;
    dw5_kernel<<<(unsigned)((tot + 255) / 256), 256, 0, stream>>>(ms, w_dw5, agg1);
  }
  // 3) grouped 1x1 -> ms[768:1536]
  {
    long tot = B * 768 * N;
    grouped_pw_kernel<<<(unsigned)((tot + 255) / 256), 256, 0, stream>>>(agg1, w_pw_g, ms);
  }

  // 4) VQ codes for q (part=0) and k (part=1)
  vq_codes_kernel<<<dim3(16, 256), 128, 0, stream>>>(ms, cb, cn2, idxq, 0);
  vq_codes_kernel<<<dim3(16, 256), 128, 0, stream>>>(ms, cb, cn2, idxk, 1);

  // 5) scatter-mean of v by k-code, gather by q-code
  (void)hipMemsetAsync(sums, 0, (size_t)B * 16 * 256 * 32 * 4, stream);
  (void)hipMemsetAsync(counts, 0, (size_t)B * 16 * 256 * 4, stream);
  {
    long tot = B * 16 * N;
    vq_scatter_kernel<<<(unsigned)((tot + 255) / 256), 256, 0, stream>>>(ms, idxk, sums, counts);
    vq_gather_kernel<<<(unsigned)((tot + 255) / 256), 256, 0, stream>>>(idxq, sums, counts, vqout);
  }

  // 6) x1 = x + bn1(w_proj @ vqout)
  gemm_wmma<3><<<dim3(16, 4, 16), 128, 0, stream>>>(
      w_proj_bf, vqout, x1, 256, 1024, 512, 512L * 1024, 256L * 1024, 256L * 1024,
      nullptr, bn1_g, bn1_b, bn1_m, bn1_v, x);

  // 7) h1 = hswish(w_exp @ x1 + b_exp)
  gemm_wmma<1><<<dim3(16, 16, 16), 128, 0, stream>>>(
      w_exp_bf, x1, h1, 1024, 1024, 256, 256L * 1024, 1024L * 1024, 0,
      b_exp, nullptr, nullptr, nullptr, nullptr, nullptr);

  // 8) h2 = hswish(dw3(h1) + b_dw3)
  {
    long tot = B * 1024 * N;
    dw3_kernel<<<(unsigned)((tot + 255) / 256), 256, 0, stream>>>(h1, w_dw3, b_dw3, h2);
  }

  // 9) out = x1 + bn2(w_pw @ h2)
  gemm_wmma<3><<<dim3(16, 4, 16), 128, 0, stream>>>(
      w_pw_bf, h2, (float*)d_out, 256, 1024, 1024, 1024L * 1024, 256L * 1024,
      256L * 1024, nullptr, bn2_g, bn2_b, bn2_m, bn2_v, x1);
}
